// resnet_nonlocal_75067438399783
// MI455X (gfx1250) — matrix operand order repaired and checked by reading, not compiled, not run on hardware
//
#include <hip/hip_runtime.h>
#include <hip/hip_bf16.h>
#include <math.h>

typedef __attribute__((ext_vector_type(16))) _Float16 v16h;
typedef __attribute__((ext_vector_type(8)))  _Float16 v8h;
typedef __attribute__((ext_vector_type(8)))  float    v8f;

#define EPSBN 1e-5f

// ---------------- direct 3x3 conv (VALID), NCHW, fp32 ----------------
__global__ void conv3x3_direct_k(const float* __restrict__ in, const float* __restrict__ w,
                                 const float* __restrict__ bias, float* __restrict__ out,
                                 int B, int Cin, int Hin, int Win,
                                 int Cout, int Hout, int Wout, int stride)
{
    int idx = blockIdx.x * blockDim.x + threadIdx.x;
    int total = B * Cout * Hout * Wout;
    if (idx >= total) return;
    int x = idx % Wout; int t = idx / Wout;
    int y = t % Hout;  t /= Hout;
    int co = t % Cout; int b = t / Cout;
    float acc = bias[co];
    const float* wbase = w + (size_t)co * Cin * 9;
    int iy0 = y * stride, ix0 = x * stride;
    for (int ci = 0; ci < Cin; ++ci) {
        const float* ib = in + ((size_t)(b * Cin + ci) * Hin + iy0) * Win + ix0;
        const float* wb = wbase + ci * 9;
        #pragma unroll
        for (int ky = 0; ky < 3; ++ky)
            #pragma unroll
            for (int kx = 0; kx < 3; ++kx)
                acc += ib[ky * Win + kx] * wb[ky * 3 + kx];
    }
    out[idx] = acc;
}

// ---------------- BN batch statistics: one block per channel ----------------
__global__ void bn_stats_k(const float* __restrict__ x, float* __restrict__ mean,
                           float* __restrict__ istd, int C, int plane, int N)
{
    int c = blockIdx.x;
    __shared__ float ss[256], sq[256];
    float s = 0.f, q = 0.f;
    for (int n = 0; n < N; ++n) {
        const float* p = x + (size_t)(n * C + c) * plane;
        for (int i = threadIdx.x; i < plane; i += blockDim.x) {
            float v = p[i]; s += v; q += v * v;
        }
    }
    ss[threadIdx.x] = s; sq[threadIdx.x] = q;
    __syncthreads();
    for (int o = blockDim.x >> 1; o > 0; o >>= 1) {
        if ((int)threadIdx.x < o) {
            ss[threadIdx.x] += ss[threadIdx.x + o];
            sq[threadIdx.x] += sq[threadIdx.x + o];
        }
        __syncthreads();
    }
    if (threadIdx.x == 0) {
        float inv = 1.f / (float)(N * plane);
        float m = ss[0] * inv;
        float v = sq[0] * inv - m * m;
        if (v < 0.f) v = 0.f;
        mean[c] = m;
        istd[c] = rsqrtf(v + EPSBN);
    }
}

// mode 0: bn  1: relu(bn)  2: res + bn  3: tanh(bn + res)
__global__ void bn_apply_k(const float* __restrict__ in, const float* __restrict__ res,
                           float* __restrict__ out, const float* __restrict__ mean,
                           const float* __restrict__ istd, const float* __restrict__ gamma,
                           const float* __restrict__ beta, int C, int plane, int mode)
{
    int idx = blockIdx.x * blockDim.x + threadIdx.x;
    int total = 2 * C * plane;
    if (idx >= total) return;
    int c = (idx / plane) % C;
    float v = (in[idx] - mean[c]) * istd[c] * gamma[c] + beta[c];
    if (mode == 1)      v = fmaxf(v, 0.f);
    else if (mode == 2) v = res[idx] + v;
    else if (mode == 3) v = tanhf(v + res[idx]);
    out[idx] = v;
}

// -------- pack NCHW fp32 (B,32,126,126) -> reflect-padded NHWC f16 (B,128,128,32) --------
__global__ void pack_nhwc_k(const float* __restrict__ in, _Float16* __restrict__ out, int B)
{
    int idx = blockIdx.x * blockDim.x + threadIdx.x;
    int total = B * 128 * 128 * 32;
    if (idx >= total) return;
    int c  = idx & 31;  int t = idx >> 5;
    int xx = t & 127;   t >>= 7;
    int yy = t & 127;   int b = t >> 7;
    int sy = yy - 1; if (sy < 0) sy = -sy; if (sy > 125) sy = 250 - sy;
    int sx = xx - 1; if (sx < 0) sx = -sx; if (sx > 125) sx = 250 - sx;
    out[idx] = (_Float16)in[((size_t)(b * 32 + c) * 126 + sy) * 126 + sx];
}

// -------- pack OIHW fp32 weights (32,32,3,3) into WMMA B-register layout, f16 --------
// bp[((tap*2+ntile)*32 + lane)*16 + e]; n = ntile*16+(lane&15); k = (e>>3)*16 + ((lane>>4)&1)*8 + (e&7)
__global__ void pack_wB_k(const float* __restrict__ w, _Float16* __restrict__ bp)
{
    int idx = blockIdx.x * blockDim.x + threadIdx.x;
    if (idx >= 9216) return;
    int e = idx & 15;          int t = idx >> 4;
    int lane = t & 31;         t >>= 5;
    int nt = t & 1;            int tap = t >> 1;
    int n = nt * 16 + (lane & 15);
    int k = (e >> 3) * 16 + ((lane >> 4) & 1) * 8 + (e & 7);
    int ky = tap / 3, kx = tap % 3;
    bp[idx] = (_Float16)w[((size_t)(n * 32 + k) * 3 + ky) * 3 + kx];
}

// -------- implicit-GEMM 3x3 conv via v_wmma_f32_16x16x32_f16 --------
// One wave: 16 output positions x 32 output channels; K = 9 taps * 32 cin.
__global__ void wmma_conv3x3_k(const _Float16* __restrict__ pin, const _Float16* __restrict__ bp,
                               const float* __restrict__ bias, float* __restrict__ out, int Mtot)
{
    const int HW = 126 * 126;
    int wave = (int)((blockIdx.x * blockDim.x + threadIdx.x) >> 5);
    int lane = threadIdx.x & 31;
    int m0 = wave * 16;
    int m = m0 + (lane & 15);
    int mv = (m < Mtot) ? m : 0;
    int b = mv / HW; int rem = mv % HW;
    int y = rem / 126, x = rem % 126;
    int kb = (lane & 16) ? 8 : 0;

    v8f c0 = {}; v8f c1 = {};
    #pragma unroll
    for (int tap = 0; tap < 9; ++tap) {
        int ky = tap / 3, kx = tap % 3;
        const _Float16* pa = pin + ((size_t)(b * 128 + (y + ky)) * 128 + (x + kx)) * 32;
        v8h lo = *(const v8h*)(pa + kb);        // K = kb..kb+7
        v8h hi = *(const v8h*)(pa + 16 + kb);   // K = 16+kb..16+kb+7
        v16h a;
        #pragma unroll
        for (int e = 0; e < 8; ++e) { a[e] = lo[e]; a[8 + e] = hi[e]; }
        v16h b0 = *(const v16h*)(bp + (size_t)((tap * 2 + 0) * 32 + lane) * 16);
        v16h b1 = *(const v16h*)(bp + (size_t)((tap * 2 + 1) * 32 + lane) * 16);
        c0 = __builtin_amdgcn_wmma_f32_16x16x32_f16(false, a, false, b0, (short)0, c0, false, false);
        c1 = __builtin_amdgcn_wmma_f32_16x16x32_f16(false, a, false, b1, (short)0, c1, false, false);
    }

    int n0 = lane & 15;
    int n1 = 16 + n0;
    float bi0 = bias[n0], bi1 = bias[n1];
    int mbase = m0 + ((lane & 16) ? 8 : 0);
    #pragma unroll
    for (int r = 0; r < 8; ++r) {
        int mo = mbase + r;
        if (mo < Mtot) {
            int bb = mo / HW; int rr = mo % HW;
            out[(size_t)(bb * 32 + n0) * HW + rr] = c0[r] + bi0;
            out[(size_t)(bb * 32 + n1) * HW + rr] = c1[r] + bi1;
        }
    }
}

// ---------------- 1x1 conv, NCHW ----------------
__global__ void conv1x1_k(const float* __restrict__ in, const float* __restrict__ w,
                          const float* __restrict__ bias, float* __restrict__ out,
                          int B, int Cin, int Cout, int plane)
{
    int idx = blockIdx.x * blockDim.x + threadIdx.x;
    int total = B * Cout * plane;
    if (idx >= total) return;
    int j = idx % plane; int t = idx / plane;
    int co = t % Cout;   int b = t / Cout;
    float acc = bias[co];
    for (int ci = 0; ci < Cin; ++ci)
        acc += in[((size_t)(b * Cin + ci)) * plane + j] * w[co * Cin + ci];
    out[idx] = acc;
}

// ---------------- fused 1x1 conv (4->2) + 2x2 maxpool: (B,4,126,126) -> (B,2,63,63) ----------------
__global__ void conv1x1_pool_k(const float* __restrict__ in, const float* __restrict__ w,
                               const float* __restrict__ bias, float* __restrict__ out, int B)
{
    int idx = blockIdx.x * blockDim.x + threadIdx.x;
    int total = B * 2 * 63 * 63;
    if (idx >= total) return;
    int x = idx % 63; int t = idx / 63;
    int y = t % 63;   t /= 63;
    int co = t % 2;   int b = t / 2;
    float m = -3.4e38f;
    #pragma unroll
    for (int dy = 0; dy < 2; ++dy)
        #pragma unroll
        for (int dx = 0; dx < 2; ++dx) {
            int yy = 2 * y + dy, xx = 2 * x + dx;
            float v = bias[co];
            #pragma unroll
            for (int ci = 0; ci < 4; ++ci)
                v += in[((size_t)(b * 4 + ci) * 126 + yy) * 126 + xx] * w[co * 4 + ci];
            m = fmaxf(m, v);
        }
    out[idx] = m;
}

// ---------------- fused attention: softmax(theta^T phi) @ g^T, online softmax ----------------
__global__ void attn_k(const float* __restrict__ theta, const float* __restrict__ phi,
                       const float* __restrict__ g, float* __restrict__ y)
{
    const int M = 3969, Nn = 15876;
    int b = blockIdx.y;
    __shared__ float p0[3969], p1[3969], g0[3969], g1[3969];
    for (int i = threadIdx.x; i < M; i += blockDim.x) {
        p0[i] = phi[(size_t)(b * 2 + 0) * M + i];
        p1[i] = phi[(size_t)(b * 2 + 1) * M + i];
        g0[i] = g[(size_t)(b * 2 + 0) * M + i];
        g1[i] = g[(size_t)(b * 2 + 1) * M + i];
    }
    __syncthreads();
    int n = blockIdx.x * blockDim.x + threadIdx.x;
    if (n >= Nn) return;
    float t0 = theta[(size_t)(b * 2 + 0) * Nn + n];
    float t1 = theta[(size_t)(b * 2 + 1) * Nn + n];
    float mx = -3.4e38f, s = 0.f, a0 = 0.f, a1 = 0.f;
    for (int j = 0; j < M; ++j) {
        float f = t0 * p0[j] + t1 * p1[j];
        float e;
        if (f > mx) {
            float sc = __expf(mx - f);
            s *= sc; a0 *= sc; a1 *= sc;
            mx = f; e = 1.f;
        } else {
            e = __expf(f - mx);
        }
        s += e; a0 += e * g0[j]; a1 += e * g1[j];
    }
    float inv = 1.f / s;
    y[(size_t)(b * 2 + 0) * Nn + n] = a0 * inv;
    y[(size_t)(b * 2 + 1) * Nn + n] = a1 * inv;
}

// ---------------- head: (2,63504) @ w1^T (256,63504) — bandwidth bound, share w1 across batch ----------------
__global__ void head1_k(const float* __restrict__ z, const float* __restrict__ w1,
                        const float* __restrict__ b1, float* __restrict__ h1)
{
    int o = blockIdx.x; // 0..255
    const float* wr = w1 + (size_t)o * 63504;
    float a0 = 0.f, a1 = 0.f;
    for (int j = threadIdx.x; j < 63504; j += blockDim.x) {
        float wv = wr[j];
        a0 += wv * z[j];
        a1 += wv * z[63504 + j];
    }
    __shared__ float s0[256], s1[256];
    s0[threadIdx.x] = a0; s1[threadIdx.x] = a1;
    __syncthreads();
    for (int off = blockDim.x >> 1; off > 0; off >>= 1) {
        if ((int)threadIdx.x < off) {
            s0[threadIdx.x] += s0[threadIdx.x + off];
            s1[threadIdx.x] += s1[threadIdx.x + off];
        }
        __syncthreads();
    }
    if (threadIdx.x == 0) {
        h1[o]       = s0[0] + b1[o];
        h1[256 + o] = s1[0] + b1[o];
    }
}

__global__ void head2_k(const float* __restrict__ h1, const float* __restrict__ w2,
                        const float* __restrict__ b2, float* __restrict__ h2)
{
    int idx = threadIdx.x; // 256 = 2*128
    int b = idx >> 7, o = idx & 127;
    float acc = b2[o];
    for (int j = 0; j < 256; ++j) acc += h1[b * 256 + j] * w2[o * 256 + j];
    h2[b * 128 + o] = acc;
}

__global__ void head3_k(const float* __restrict__ h2, const float* __restrict__ wo,
                        const float* __restrict__ bo, float* __restrict__ out)
{
    int idx = threadIdx.x;
    if (idx >= 20) return;
    int b = idx / 10, o = idx % 10;
    float acc = bo[o];
    for (int j = 0; j < 128; ++j) acc += h2[b * 128 + j] * wo[o * 128 + j];
    out[b * 10 + o] = acc;
}

extern "C" void kernel_launch(void* const* d_in, const int* in_sizes, int n_in,
                              void* d_out, int out_size, void* d_ws, size_t ws_size,
                              hipStream_t stream) {
    (void)in_sizes; (void)n_in; (void)out_size; (void)ws_size;
    const float* X = (const float*)d_in[0];
    auto F = [&](int i) { return (const float*)d_in[i]; };

    // ---- workspace layout (fp32 units, 256B aligned chunks) ----
    float* wsf = (float*)d_ws;
    size_t off = 0;
    auto alloc = [&](size_t n) { float* p = wsf + off; off += (n + 63) & ~(size_t)63; return p; };
    const int S1 = 2 * 16 * 254 * 254;      // 2,064,512
    const int S2 = 2 * 32 * 126 * 126;      // 1,016,064
    float*     f_conv1 = alloc(S1);
    float*     f_act1  = alloc(S1);
    float*     f_c2    = alloc(S2);
    float*     f_x     = alloc(S2);
    float*     f_t1    = alloc(S2);
    float*     f_t2    = alloc(S2);
    _Float16*  padb    = (_Float16*)alloc(2 * 128 * 128 * 32 / 2);
    _Float16*  bpk     = (_Float16*)alloc(9216 / 2 + 64);
    float*     f_xloc  = alloc(2 * 4 * 15876);
    float*     f_xnew  = alloc(2 * 4 * 15876);
    float*     f_theta = alloc(2 * 2 * 15876);
    float*     f_g     = alloc(2 * 2 * 3969);
    float*     f_phi   = alloc(2 * 2 * 3969);
    float*     f_y     = alloc(2 * 2 * 15876);
    float*     f_wy    = alloc(2 * 4 * 15876);
    float*     f_z     = alloc(2 * 4 * 15876);
    float*     f_h1    = alloc(512);
    float*     f_h2    = alloc(256);
    float*     f_mean  = alloc(32);
    float*     f_istd  = alloc(32);

    auto bn = [&](const float* in, int C, int plane, const float* gm, const float* bt,
                  const float* res, float* outp, int mode) {
        bn_stats_k<<<C, 256, 0, stream>>>(in, f_mean, f_istd, C, plane, 2);
        int tot = 2 * C * plane;
        bn_apply_k<<<(tot + 255) / 256, 256, 0, stream>>>(in, res, outp, f_mean, f_istd, gm, bt, C, plane, mode);
    };

    const int Mtot = 2 * 15876;
    const int wmma_blocks = ((Mtot + 15) / 16 + 3) / 4; // 4 waves/block, 16 rows/wave
    const int packN = 2 * 128 * 128 * 32;

    // ---- encoder: indices base..base+25 (w1,b1,g1,be1,w2,b2,g2,be2,wo,bo,2x blocks) ----
    auto run_enc = [&](int base, float* out4) {
        conv3x3_direct_k<<<(S1 + 255) / 256, 256, 0, stream>>>(
            X, F(base + 0), F(base + 1), f_conv1, 2, 3, 256, 256, 16, 254, 254, 1);
        bn(f_conv1, 16, 254 * 254, F(base + 2), F(base + 3), nullptr, f_act1, 1);
        conv3x3_direct_k<<<(S2 + 255) / 256, 256, 0, stream>>>(
            f_act1, F(base + 4), F(base + 5), f_c2, 2, 16, 254, 254, 32, 126, 126, 2);
        bn(f_c2, 32, 15876, F(base + 6), F(base + 7), nullptr, f_x, 1);
        for (int blk = 0; blk < 2; ++blk) {
            int bb = base + 10 + blk * 8;
            // h = relu(bn(conv(rpad(x))))
            pack_nhwc_k<<<(packN + 255) / 256, 256, 0, stream>>>(f_x, padb, 2);
            pack_wB_k<<<(9216 + 255) / 256, 256, 0, stream>>>(F(bb + 0), bpk);
            wmma_conv3x3_k<<<wmma_blocks, 128, 0, stream>>>(padb, bpk, F(bb + 1), f_t1, Mtot);
            bn(f_t1, 32, 15876, F(bb + 2), F(bb + 3), nullptr, f_t2, 1);
            // h2 = bn(conv(rpad(h)));  x = x + h2
            pack_nhwc_k<<<(packN + 255) / 256, 256, 0, stream>>>(f_t2, padb, 2);
            pack_wB_k<<<(9216 + 255) / 256, 256, 0, stream>>>(F(bb + 4), bpk);
            wmma_conv3x3_k<<<wmma_blocks, 128, 0, stream>>>(padb, bpk, F(bb + 5), f_t1, Mtot);
            bn(f_t1, 32, 15876, F(bb + 6), F(bb + 7), f_x, f_x, 2);
        }
        conv1x1_k<<<(2 * 4 * 15876 + 255) / 256, 256, 0, stream>>>(
            f_x, F(base + 8), F(base + 9), out4, 2, 32, 4, 15876);
    };

    run_enc(1,  f_xloc);   // enc_local
    run_enc(27, f_xnew);   // enc_new

    // ---- non-local block: loc = x_new, glb = x_local ----
    conv1x1_k<<<(2 * 2 * 15876 + 255) / 256, 256, 0, stream>>>(
        f_xnew, F(55), F(56), f_theta, 2, 4, 2, 15876);                       // theta
    conv1x1_pool_k<<<(2 * 2 * 63 * 63 + 255) / 256, 256, 0, stream>>>(f_xloc, F(53), F(54), f_g, 2);   // g
    conv1x1_pool_k<<<(2 * 2 * 63 * 63 + 255) / 256, 256, 0, stream>>>(f_xloc, F(57), F(58), f_phi, 2); // phi
    attn_k<<<dim3(63, 2), 256, 0, stream>>>(f_theta, f_phi, f_g, f_y);
    conv1x1_k<<<(2 * 4 * 15876 + 255) / 256, 256, 0, stream>>>(
        f_y, F(59), F(60), f_wy, 2, 2, 4, 15876);                             // W(y)
    bn(f_wy, 4, 15876, F(61), F(62), f_xnew, f_z, 3);                          // z = tanh(bn + loc)

    // ---- head ----
    head1_k<<<256, 256, 0, stream>>>(f_z, F(63), F(64), f_h1);
    head2_k<<<1, 256, 0, stream>>>(f_h1, F(65), F(66), f_h2);
    head3_k<<<1, 32, 0, stream>>>(f_h2, F(67), F(68), (float*)d_out);
}
